// SE3TransformerLayer_16423954939934
// MI455X (gfx1250) — matrix operand order repaired and checked by reading, not compiled, not run on hardware
//
#include <hip/hip_runtime.h>
#include <cstdint>
#include <cstddef>

// ---------------------------------------------------------------------------
// Types for CDNA5 WMMA (wave32): V_WMMA_F32_16X16X32_BF16
// ---------------------------------------------------------------------------
typedef __attribute__((ext_vector_type(16))) __bf16 v16bf;
typedef __attribute__((ext_vector_type(8)))  __bf16 v8bf;
typedef __attribute__((ext_vector_type(8)))  float  v8f;

constexpr int Bb = 4;
constexpr int Nn = 2048;
constexpr int Hh = 512;
constexpr int FFd = 2048;
constexpr int Mm = Bb * Nn;       // 8192 total token rows
constexpr float SCALE = 0.04419417382415922f;  // 1/sqrt(512)

__device__ __forceinline__ v8f v8f_zero() {
  v8f z = {0.f, 0.f, 0.f, 0.f, 0.f, 0.f, 0.f, 0.f};
  return z;
}

__device__ __forceinline__ v16bf cat8(v8bf lo, v8bf hi) {
  return __builtin_shufflevector(lo, hi, 0, 1, 2, 3, 4, 5, 6, 7,
                                         8, 9, 10, 11, 12, 13, 14, 15);
}

// A-operand fragment (16x32 bf16, row-major source). Caller passes
// p = rowBase + 8*lhalf; lanes 0-15 take K {0..7,16..23}, lanes 16-31 {8..15,24..31}.
__device__ __forceinline__ v16bf ldA(const __bf16* p) {
  v8bf lo = *(const v8bf*)p;
  v8bf hi = *(const v8bf*)(p + 16);
  return cat8(lo, hi);
}

// B-operand fragment (32x16 bf16, K-major source). Caller passes
// p = colBase + 8*lhalf; lanes 0-15 take K {0..7,16..23}, lanes 16-31 {8..15,24..31}.
__device__ __forceinline__ v16bf ldB16(const __bf16* p) {
  v8bf lo = *(const v8bf*)p;
  v8bf hi = *(const v8bf*)(p + 16);
  return cat8(lo, hi);
}

__device__ __forceinline__ v8f wmma_bf16(v16bf a, v16bf b, v8f c) {
  return __builtin_amdgcn_wmma_f32_16x16x32_bf16(
      /*neg_a=*/false, a, /*neg_b=*/false, b,
      /*c_mod=*/(short)0, c, /*reuse_a=*/false, /*reuse_b=*/false);
}

// ---------------------------------------------------------------------------
// Pack kernels: f32 -> bf16 (row-major copy), and f32 [K][N] -> bf16 [N][K]
// ---------------------------------------------------------------------------
__global__ void pack_bf16_kernel(const float* __restrict__ src,
                                 __bf16* __restrict__ dst, int n8) {
  int i = blockIdx.x * blockDim.x + threadIdx.x;
  if (i < n8) {
    const float* p = src + (size_t)i * 8;
    v8bf o;
#pragma unroll
    for (int e = 0; e < 8; ++e) o[e] = (__bf16)p[e];
    *(v8bf*)(dst + (size_t)i * 8) = o;
  }
}

__global__ void transpose_bf16_kernel(const float* __restrict__ w,
                                      __bf16* __restrict__ wT, int K, int N) {
  int i = blockIdx.x * blockDim.x + threadIdx.x;  // N * (K/8) work items
  int kc = i % (K / 8);
  int n  = i / (K / 8);
  if (n < N) {
    v8bf o;
#pragma unroll
    for (int e = 0; e < 8; ++e)
      o[e] = (__bf16)w[(size_t)(kc * 8 + e) * N + n];
    *(v8bf*)(wT + (size_t)n * K + kc * 8) = o;
  }
}

// ---------------------------------------------------------------------------
// Generic WMMA GEMM: C[Mm x N] = A[Mm x K](bf16,row-major) @ Bt[N x K](bf16,K-major)
// 256 threads (8 waves); WG tile 128x128; wave tile 64x32 (4x2 accum tiles).
// MODE 0: QKV  -> Q bf16 [M][512], K bf16 [M][512], V^T bf16 [B][512][2048], +bias
// MODE 1: outF = (C + bias) * mask                          (out-projection)
// MODE 2: outQ(bf16) = relu((C + bias) * mask)              (FFN layer 1)
// MODE 3: outF = relu((C + bias) * mask)                    (FFN layer 2)
// ---------------------------------------------------------------------------
template <int MODE, int N, int K>
__global__ __launch_bounds__(256) void gemm_wmma(
    const __bf16* __restrict__ A, const __bf16* __restrict__ Bt,
    const float* __restrict__ bias, const unsigned char* __restrict__ mask,
    float* __restrict__ outF, __bf16* __restrict__ outQ,
    __bf16* __restrict__ outK, __bf16* __restrict__ outVT) {
  const int m0 = blockIdx.x * 128;
  const int n0 = blockIdx.y * 128;
  const int tid = threadIdx.x;
  const int w = tid >> 5, lane = tid & 31;
  const int l15 = lane & 15, lhalf = lane >> 4;
  const int mbase = m0 + (w >> 2) * 64;
  const int nbase = n0 + (w & 3) * 32;

  v8f acc[4][2];
#pragma unroll
  for (int i = 0; i < 4; ++i)
#pragma unroll
    for (int j = 0; j < 2; ++j) acc[i][j] = v8f_zero();

  const __bf16* aBase = A + (size_t)(mbase + l15) * K + 8 * lhalf;
  const __bf16* bBase = Bt + (size_t)(nbase + l15) * K + 8 * lhalf;

#pragma unroll 4
  for (int k0 = 0; k0 < K; k0 += 32) {
    v16bf af[4], bfr[2];
#pragma unroll
    for (int i = 0; i < 4; ++i) af[i] = ldA(aBase + (size_t)i * 16 * K + k0);
#pragma unroll
    for (int j = 0; j < 2; ++j) bfr[j] = ldB16(bBase + (size_t)j * 16 * K + k0);
#pragma unroll
    for (int i = 0; i < 4; ++i)
#pragma unroll
      for (int j = 0; j < 2; ++j)
        acc[i][j] = wmma_bf16(af[i], bfr[j], acc[i][j]);
  }

  // Epilogue. D layout: lane l15 = column; rows = mrow0 + r (r=0..7),
  // mrow0 = tile_m + 8*lhalf.
#pragma unroll
  for (int i = 0; i < 4; ++i) {
    const int mrow0 = mbase + i * 16 + 8 * lhalf;
    float mk[8];
    if (MODE >= 1) {
#pragma unroll
      for (int r = 0; r < 8; ++r) mk[r] = mask[mrow0 + r] ? 1.0f : 0.0f;
    }
#pragma unroll
    for (int j = 0; j < 2; ++j) {
      const int n = nbase + j * 16 + l15;
      const float bv = bias[n];
      if (MODE == 0) {
        const int sec = n >> 9;      // 0 = Q, 1 = K, 2 = V
        const int nl = n & 511;
        if (sec == 2) {
          // V stored transposed [B][H][N]; rows of tile = consecutive tokens.
          v8bf pk;
#pragma unroll
          for (int r = 0; r < 8; ++r) pk[r] = (__bf16)(acc[i][j][r] + bv);
          const int bidx = mrow0 >> 11;
          const int tok = mrow0 & 2047;
          *(v8bf*)(outVT + ((size_t)(bidx * 512 + nl)) * 2048 + tok) = pk;
        } else {
          __bf16* dst = (sec == 0) ? outQ : outK;
#pragma unroll
          for (int r = 0; r < 8; ++r)
            dst[(size_t)(mrow0 + r) * 512 + nl] = (__bf16)(acc[i][j][r] + bv);
        }
      } else if (MODE == 1) {
#pragma unroll
        for (int r = 0; r < 8; ++r)
          outF[(size_t)(mrow0 + r) * N + n] = (acc[i][j][r] + bv) * mk[r];
      } else if (MODE == 2) {
#pragma unroll
        for (int r = 0; r < 8; ++r) {
          float z = (acc[i][j][r] + bv) * mk[r];
          outQ[(size_t)(mrow0 + r) * N + n] = (__bf16)fmaxf(z, 0.f);
        }
      } else {
#pragma unroll
        for (int r = 0; r < 8; ++r) {
          float z = (acc[i][j][r] + bv) * mk[r];
          outF[(size_t)(mrow0 + r) * N + n] = fmaxf(z, 0.f);
        }
      }
    }
  }
}

// ---------------------------------------------------------------------------
// Flash-style attention with geometric distance modulation.
// One WG = (batch b, 16 query rows). 8 waves; wave w owns H-slice [64w,64w+64).
// 4 rounds of 512 keys; S stripe lives in LDS (16 x 516 f32 ~ 33 KB).
// ---------------------------------------------------------------------------
__global__ __launch_bounds__(256) void attn_kernel(
    const __bf16* __restrict__ Qbf, const __bf16* __restrict__ Kbf,
    const __bf16* __restrict__ VTbf, const float* __restrict__ geom,
    const unsigned char* __restrict__ mask, __bf16* __restrict__ ctxbf) {
  constexpr int RS = 516;  // padded stripe row stride (floats)
  __shared__ float S[16 * RS];
  __shared__ float red[16][16];
  __shared__ float rowmax[16], rowsum[16], rowalpha[16];
  __shared__ float pq[16][4];

  const int b = blockIdx.x >> 7;
  const int m0 = (blockIdx.x & 127) << 4;
  const int tid = threadIdx.x;
  const int w = tid >> 5, lane = tid & 31;
  const int l15 = lane & 15, lhalf = lane >> 4;

  const __bf16* Qb = Qbf + ((size_t)(b * 2048 + m0)) * 512;
  const __bf16* Kb = Kbf + (size_t)b * 2048 * 512;
  const __bf16* Vb = VTbf + (size_t)b * 512 * 2048;
  const float* gb = geom + (size_t)b * 2048 * 3;
  const unsigned char* mb = mask + (size_t)b * 2048;

  if (tid < 16) {
    float px = gb[(m0 + tid) * 3 + 0];
    float py = gb[(m0 + tid) * 3 + 1];
    float pz = gb[(m0 + tid) * 3 + 2];
    pq[tid][0] = px; pq[tid][1] = py; pq[tid][2] = pz;
    pq[tid][3] = px * px + py * py + pz * pz;
    rowmax[tid] = -1e30f;
    rowsum[tid] = 0.f;
  }
  __syncthreads();

  const __bf16* ap0 = Qb + l15 * 512 + 8 * lhalf;
  const int h0 = w << 6;
  v8f o[4];
#pragma unroll
  for (int u = 0; u < 4; ++u) o[u] = v8f_zero();

  for (int R = 0; R < 4; ++R) {
    const int K0 = R << 9;
    // ---- Phase A: S tiles (Q @ K^T), distance/mask/scale, write stripe ----
    for (int t = 0; t < 4; ++t) {
      const int jloc = ((t << 3) + w) << 4;  // 0..511, stripes across waves
      const int j0 = K0 + jloc;
      v8f acc = v8f_zero();
      const __bf16* bp0 = Kb + (size_t)(j0 + l15) * 512 + 8 * lhalf;
#pragma unroll 4
      for (int k0 = 0; k0 < 512; k0 += 32)
        acc = wmma_bf16(ldA(ap0 + k0), ldB16(bp0 + k0), acc);

      const int j = j0 + l15;  // this lane's key (column)
      const float kx = gb[j * 3 + 0];
      const float ky = gb[j * 3 + 1];
      const float kz = gb[j * 3 + 2];
      const float sqk = kx * kx + ky * ky + kz * kz;
      const bool mj = mb[j] != 0;
#pragma unroll
      for (int r = 0; r < 8; ++r) {
        const int lr = r + (lhalf << 3);
        float d2 = pq[lr][3] + sqk -
                   2.f * (pq[lr][0] * kx + pq[lr][1] * ky + pq[lr][2] * kz);
        float dist = sqrtf(fmaxf(d2, 1e-12f));
        float s = acc[r] * SCALE * __expf(-dist);
        S[lr * RS + jloc + l15] = mj ? s : -1e30f;
      }
    }
    __syncthreads();

    // ---- Phase B: online softmax update over this 512-key round ----
    {
      const int row = tid >> 4, c0 = tid & 15;
      float mh = -1e30f;
      for (int c = c0; c < 512; c += 16) mh = fmaxf(mh, S[row * RS + c]);
      red[row][c0] = mh;
      __syncthreads();
      if (c0 == 0) {
        float m2 = -1e30f;
#pragma unroll
        for (int i = 0; i < 16; ++i) m2 = fmaxf(m2, red[row][i]);
        float nm = fmaxf(rowmax[row], m2);
        rowalpha[row] = __expf(rowmax[row] - nm);
        rowmax[row] = nm;
      }
      __syncthreads();
      const float nm = rowmax[row];
      float ps = 0.f;
      for (int c = c0; c < 512; c += 16) {
        float e = __expf(S[row * RS + c] - nm);
        S[row * RS + c] = e;
        ps += e;
      }
      red[row][c0] = ps;
      __syncthreads();
      if (c0 == 0) {
        float s2 = 0.f;
#pragma unroll
        for (int i = 0; i < 16; ++i) s2 += red[row][i];
        rowsum[row] = rowsum[row] * rowalpha[row] + s2;
      }
      __syncthreads();
    }

    // ---- Phase C: O = O*alpha + P @ V  (wave-private 64-wide H slice) ----
    {
#pragma unroll
      for (int u = 0; u < 4; ++u)
#pragma unroll
        for (int r = 0; r < 8; ++r) o[u][r] *= rowalpha[r + (lhalf << 3)];

      for (int jl = 0; jl < 512; jl += 32) {
        v16bf a;
        const float* sp = &S[l15 * RS + jl + 8 * lhalf];
#pragma unroll
        for (int e = 0; e < 8; ++e) {
          a[e] = (__bf16)sp[e];
          a[8 + e] = (__bf16)sp[16 + e];
        }
#pragma unroll
        for (int u = 0; u < 4; ++u) {
          const __bf16* vp =
              Vb + (size_t)(h0 + (u << 4) + l15) * 2048 + K0 + jl + 8 * lhalf;
          o[u] = wmma_bf16(a, ldB16(vp), o[u]);
        }
      }
    }
    __syncthreads();  // stripe reused next round
  }

  // ---- Normalize by rowsum, store context as bf16 [B][N][H] ----
  float rinv[8];
#pragma unroll
  for (int r = 0; r < 8; ++r) rinv[r] = 1.0f / rowsum[r + (lhalf << 3)];
#pragma unroll
  for (int u = 0; u < 4; ++u) {
    const int n = h0 + (u << 4) + l15;
#pragma unroll
    for (int r = 0; r < 8; ++r) {
      const int lr = r + (lhalf << 3);
      ctxbf[((size_t)(b * 2048 + m0 + lr)) * 512 + n] = (__bf16)(o[u][r] * rinv[r]);
    }
  }
}

// ---------------------------------------------------------------------------
// LayerNorm over H=512: out = (a+res - mu) * rsqrt(var+eps) * g + b
// One 256-thread WG per row; optional bf16 mirror output.
// ---------------------------------------------------------------------------
__global__ __launch_bounds__(256) void ln_kernel(
    const float* __restrict__ a, const float* __restrict__ res,
    const float* __restrict__ g, const float* __restrict__ bb,
    float* __restrict__ outF, __bf16* __restrict__ outBf) {
  __shared__ float wsum[8];
  __shared__ float wsq[8];
  const int row = blockIdx.x;
  const int tid = threadIdx.x;
  const int w = tid >> 5, lane = tid & 31;
  const size_t base = (size_t)row * 512;

  float e0 = a[base + tid] + res[base + tid];
  float e1 = a[base + tid + 256] + res[base + tid + 256];

  float s = e0 + e1;
#pragma unroll
  for (int off = 16; off > 0; off >>= 1) s += __shfl_xor(s, off, 32);
  if (lane == 0) wsum[w] = s;
  __syncthreads();
  float tot = 0.f;
#pragma unroll
  for (int i = 0; i < 8; ++i) tot += wsum[i];
  const float mu = tot * (1.0f / 512.0f);

  const float d0 = e0 - mu, d1 = e1 - mu;
  float q = d0 * d0 + d1 * d1;
#pragma unroll
  for (int off = 16; off > 0; off >>= 1) q += __shfl_xor(q, off, 32);
  if (lane == 0) wsq[w] = q;
  __syncthreads();
  float vt = 0.f;
#pragma unroll
  for (int i = 0; i < 8; ++i) vt += wsq[i];
  const float rstd = rsqrtf(vt * (1.0f / 512.0f) + 1e-5f);

  const float y0 = d0 * rstd * g[tid] + bb[tid];
  const float y1 = d1 * rstd * g[tid + 256] + bb[tid + 256];
  outF[base + tid] = y0;
  outF[base + tid + 256] = y1;
  if (outBf != nullptr) {
    outBf[base + tid] = (__bf16)y0;
    outBf[base + tid + 256] = (__bf16)y1;
  }
}

// ---------------------------------------------------------------------------
// Host orchestration
// ---------------------------------------------------------------------------
static inline int cdiv_i(int a, int b) { return (a + b - 1) / b; }

extern "C" void kernel_launch(void* const* d_in, const int* in_sizes, int n_in,
                              void* d_out, int out_size, void* d_ws,
                              size_t ws_size, hipStream_t stream) {
  (void)in_sizes; (void)n_in; (void)out_size; (void)ws_size;

  const float* x      = (const float*)d_in[0];
  const float* geom   = (const float*)d_in[1];
  const unsigned char* mask = (const unsigned char*)d_in[2];
  const float* w_qkv  = (const float*)d_in[3];
  const float* b_qkv  = (const float*)d_in[4];
  const float* w_out  = (const float*)d_in[5];
  const float* b_out  = (const float*)d_in[6];
  const float* w1     = (const float*)d_in[7];
  const float* b1     = (const float*)d_in[8];
  const float* w2     = (const float*)d_in[9];
  const float* b2     = (const float*)d_in[10];
  const float* ln1_g  = (const float*)d_in[11];
  const float* ln1_b  = (const float*)d_in[12];
  const float* ln2_g  = (const float*)d_in[13];
  const float* ln2_b  = (const float*)d_in[14];

  char* ws = (char*)d_ws;
  size_t off = 0;
  auto take = [&](size_t bytes) -> char* {
    char* p = ws + off;
    off += (bytes + 255) & ~(size_t)255;
    return p;
  };
  __bf16* xbf   = (__bf16*)take((size_t)Mm * Hh * 2);
  __bf16* wqkvT = (__bf16*)take((size_t)3 * Hh * Hh * 2);
  __bf16* woutT = (__bf16*)take((size_t)Hh * Hh * 2);
  __bf16* w1T   = (__bf16*)take((size_t)FFd * Hh * 2);
  __bf16* w2T   = (__bf16*)take((size_t)Hh * FFd * 2);
  __bf16* Qbf   = (__bf16*)take((size_t)Mm * Hh * 2);
  __bf16* Kbf   = (__bf16*)take((size_t)Mm * Hh * 2);
  __bf16* VTbf  = (__bf16*)take((size_t)Mm * Hh * 2);
  __bf16* ctxbf = (__bf16*)take((size_t)Mm * Hh * 2);
  float*  attnF = (float*)take((size_t)Mm * Hh * 4);
  float*  x1F   = (float*)take((size_t)Mm * Hh * 4);
  __bf16* x1bf  = (__bf16*)take((size_t)Mm * Hh * 2);
  __bf16* hbf   = (__bf16*)take((size_t)Mm * FFd * 2);
  float*  ffnF  = (float*)take((size_t)Mm * Hh * 4);

  // 1. Pack activations + weights to bf16 (weights K-major transposed).
  {
    int n8 = Mm * Hh / 8;
    pack_bf16_kernel<<<cdiv_i(n8, 256), 256, 0, stream>>>(x, xbf, n8);
    int t;
    t = (3 * Hh) * (Hh / 8);
    transpose_bf16_kernel<<<cdiv_i(t, 256), 256, 0, stream>>>(w_qkv, wqkvT, Hh, 3 * Hh);
    t = Hh * (Hh / 8);
    transpose_bf16_kernel<<<cdiv_i(t, 256), 256, 0, stream>>>(w_out, woutT, Hh, Hh);
    t = FFd * (Hh / 8);
    transpose_bf16_kernel<<<cdiv_i(t, 256), 256, 0, stream>>>(w1, w1T, Hh, FFd);
    t = Hh * (FFd / 8);
    transpose_bf16_kernel<<<cdiv_i(t, 256), 256, 0, stream>>>(w2, w2T, FFd, Hh);
  }

  // 2. QKV projection (writes Q,K row-major bf16 and V transposed bf16).
  gemm_wmma<0, 3 * Hh, Hh><<<dim3(Mm / 128, (3 * Hh) / 128), 256, 0, stream>>>(
      xbf, wqkvT, b_qkv, nullptr, nullptr, Qbf, Kbf, VTbf);

  // 3. Distance-modulated attention with online softmax.
  attn_kernel<<<Bb * (Nn / 16), 256, 0, stream>>>(Qbf, Kbf, VTbf, geom, mask, ctxbf);

  // 4. Output projection (+bias, *mask).
  gemm_wmma<1, Hh, Hh><<<dim3(Mm / 128, Hh / 128), 256, 0, stream>>>(
      ctxbf, woutT, b_out, mask, attnF, nullptr, nullptr, nullptr);

  // 5. x1 = LN(x + attn_out)  (f32 + bf16 mirror)
  ln_kernel<<<Mm, 256, 0, stream>>>(x, attnF, ln1_g, ln1_b, x1F, x1bf);

  // 6. h = relu((x1 @ w1 + b1) * mask)  -> bf16
  gemm_wmma<2, FFd, Hh><<<dim3(Mm / 128, FFd / 128), 256, 0, stream>>>(
      x1bf, w1T, b1, mask, nullptr, hbf, nullptr, nullptr);

  // 7. ffn = relu((h @ w2 + b2) * mask) -> f32
  gemm_wmma<3, Hh, FFd><<<dim3(Mm / 128, Hh / 128), 256, 0, stream>>>(
      hbf, w2T, b2, mask, ffnF, nullptr, nullptr, nullptr);

  // 8. out = LN(x1 + ffn)
  ln_kernel<<<Mm, 256, 0, stream>>>(x1F, ffnF, ln2_g, ln2_b, (float*)d_out, nullptr);
}
